// GNNStack_26551487823974
// MI455X (gfx1250) — matrix operand order repaired and checked by reading, not compiled, not run on hardware
//
#include <hip/hip_runtime.h>
#include <hip/hip_bf16.h>

#define NNODES 50000
#define NEDGES 1600000
#define D      128
#define DOUT   3
#define EPSLN  1e-5f
#define WPACK  (8 * 4 * 32 * 16)   // packed bf16 weight elements (= 16384)

typedef __attribute__((ext_vector_type(16))) __bf16 v16bf;
typedef __attribute__((ext_vector_type(8)))  __bf16 v8bf;
typedef __attribute__((ext_vector_type(8)))  float  v8f;

// float -> bf16, round-to-nearest-even, via bit ops (no fptrunc dependence)
__device__ __forceinline__ __bf16 tobf(float f) {
    union { float f; unsigned u; } x; x.f = f;
    unsigned r = (x.u + 0x7FFFu + ((x.u >> 16) & 1u)) >> 16;
    union { unsigned short s; __bf16 b; } y; y.s = (unsigned short)r;
    return y.b;
}

// ---------------------------------------------------------------------------
// Pack a 128x128 f32 weight matrix into bf16, laid out in this code's contiguous-k per-lane
// WMMA B-fragment order: Wp[((nt*4 + kc)*32 + lane)*16 + j] where
// k = kc*32 + (lane>>4)*16 + j, n = nt*16 + (lane&15). Done ONCE per weight.
// ---------------------------------------------------------------------------
__global__ void k_pack_w(const float* __restrict__ W, __bf16* __restrict__ Wp) {
    int p = blockIdx.x * 256 + threadIdx.x;
    if (p >= WPACK) return;
    int j  = p & 15;
    int L  = (p >> 4) & 31;
    int kc = (p >> 9) & 3;
    int nt = p >> 11;
    int k  = kc * 32 + (L >> 4) * 16 + j;
    int n  = nt * 16 + (L & 15);
    Wp[p] = tobf(W[k * D + n]);
}

// ---------------------------------------------------------------------------
// WMMA GEMM: C[M x 128] = A[M x 128] @ W[128 x 128] (+ bias)
// grid.x = M/16 (= 3125 exact), block = (32, 8): 8 waves, each owns a 16x16
// N-tile. A tile staged into LDS as bf16 (row stride 136 -> conflict-free).
// Inner loop: 2x ds_load_b128 (A frag) + 32B coalesced global (B frag) + WMMA.
// EXEC is all-ones everywhere (ISA 7.12 requirement).
// ---------------------------------------------------------------------------
__global__ __launch_bounds__(256) void k_gemm_bf16(
    const float* __restrict__ A, const __bf16* __restrict__ Wp,
    const float* __restrict__ bias, float* __restrict__ C)
{
    __shared__ __bf16 atile[16 * 136];
    const int lane = threadIdx.x;            // 0..31
    const int wave = threadIdx.y;            // 0..7 (= N tile)
    const int tid  = wave * 32 + lane;
    const int m0   = blockIdx.x * 16;

    // cooperative, coalesced load + convert of the 16x128 A tile
#pragma unroll
    for (int i = 0; i < 8; ++i) {
        int idx = tid + i * 256;             // 0..2047
        int r = idx >> 7, c = idx & 127;
        atile[r * 136 + c] = tobf(A[(m0 + r) * D + c]);
    }
    __syncthreads();

    const int nl   = lane & 15;
    const int hi   = lane >> 4;              // 0 or 1
    const int ncol = wave * 16 + nl;
    const __bf16* arow = &atile[nl * 136];   // this lane's M-row of A
    v8f acc = {0.f, 0.f, 0.f, 0.f, 0.f, 0.f, 0.f, 0.f};
    union { v16bf v; v8bf h[2]; } a;

#pragma unroll
    for (int kc = 0; kc < 4; ++kc) {         // K = 128 in 4 chunks of 32
        const int k0 = kc * 32;
        // A fragment: lane's 16 halves are one contiguous run k = hi*16 + s (matches packed B)
        a.h[0] = *(const v8bf*)(arow + k0 + hi * 16);
        a.h[1] = *(const v8bf*)(arow + k0 + hi * 16 + 8);
        // B fragment: one contiguous 32-byte load from the packed weights
        v16bf b = *(const v16bf*)(Wp + (((wave * 4 + kc) * 32 + lane) << 4));
        acc = __builtin_amdgcn_wmma_f32_16x16x32_bf16(
            false, a.v, false, b, (short)0, acc, false, false);
    }

    const float bv = bias ? bias[ncol] : 0.f;
    // C/D layout: VGPR r holds M = hi*8 + r, N = lane&15
#pragma unroll
    for (int r = 0; r < 8; ++r)
        C[(m0 + hi * 8 + r) * D + ncol] = acc[r] + bv;
}

// ---------------------------------------------------------------------------
// Degree / normalization
// ---------------------------------------------------------------------------
__global__ void k_deg_init(float* __restrict__ deg) {
    int i = blockIdx.x * 256 + threadIdx.x;
    if (i < NNODES) deg[i] = 1.0f;           // self-loop contribution
}
__global__ void k_deg_accum(const int* __restrict__ col, float* __restrict__ deg) {
    int e = blockIdx.x * 256 + threadIdx.x;
    if (e < NEDGES) atomicAdd(&deg[col[e]], 1.0f);
}
__global__ void k_deg_rsqrt(float* __restrict__ deg) {
    int i = blockIdx.x * 256 + threadIdx.x;
    if (i < NNODES) deg[i] = rsqrtf(deg[i]); // deg >= 1 always
}

// ---------------------------------------------------------------------------
// Aggregation: out = bias + self-loop term, then edge scatter-add.
// ---------------------------------------------------------------------------
__global__ void k_agg_init(const float* __restrict__ h, const float* __restrict__ bias,
                           const float* __restrict__ dinv, float* __restrict__ out)
{
    int idx = blockIdx.x * 256 + threadIdx.x;    // NNODES*D = 6.4M
    if (idx < NNODES * D) {
        int node = idx >> 7, f = idx & 127;
        float di = dinv[node];
        out[idx] = bias[f] + h[idx] * di * di;
    }
}

// one wave32 per edge; float4 per lane = 128 floats/edge; f32 atomic adds.
// h (25.6 MB) is L2-resident on the 192 MB L2, so this is L2-bound, not HBM.
__global__ __launch_bounds__(256) void k_scatter(
    const float* __restrict__ h, const int* __restrict__ row,
    const int* __restrict__ col, const float* __restrict__ dinv,
    float* __restrict__ out)
{
    int e = blockIdx.x * 8 + threadIdx.y;
    if (e >= NEDGES) return;
    int r = row[e], c = col[e];
    float nrm = dinv[r] * dinv[c];
    float4 v = ((const float4*)(h + r * D))[threadIdx.x];
    float* o = out + c * D + threadIdx.x * 4;
    atomicAdd(o + 0, v.x * nrm);
    atomicAdd(o + 1, v.y * nrm);
    atomicAdd(o + 2, v.z * nrm);
    atomicAdd(o + 3, v.w * nrm);
}

// ---------------------------------------------------------------------------
// Fused ReLU + LayerNorm, one wave32 per node (wave-level shuffle reduction)
// ---------------------------------------------------------------------------
__global__ __launch_bounds__(256) void k_relu_ln(
    float* __restrict__ h, const float* __restrict__ g, const float* __restrict__ beta)
{
    int node = blockIdx.x * 8 + threadIdx.y;
    if (node >= NNODES) return;
    float4* p = (float4*)(h + node * D);
    float4 v = p[threadIdx.x];
    v.x = fmaxf(v.x, 0.f); v.y = fmaxf(v.y, 0.f);
    v.z = fmaxf(v.z, 0.f); v.w = fmaxf(v.w, 0.f);
    float s = v.x + v.y + v.z + v.w;
    float q = v.x * v.x + v.y * v.y + v.z * v.z + v.w * v.w;
#pragma unroll
    for (int off = 16; off > 0; off >>= 1) {
        s += __shfl_xor(s, off, 32);
        q += __shfl_xor(q, off, 32);
    }
    float mu  = s * (1.0f / 128.0f);
    float var = q * (1.0f / 128.0f) - mu * mu;
    float inv = rsqrtf(var + EPSLN);
    float4 gg = ((const float4*)g)[threadIdx.x];
    float4 bb = ((const float4*)beta)[threadIdx.x];
    v.x = (v.x - mu) * inv * gg.x + bb.x;
    v.y = (v.y - mu) * inv * gg.y + bb.y;
    v.z = (v.z - mu) * inv * gg.z + bb.z;
    v.w = (v.w - mu) * inv * gg.w + bb.w;
    p[threadIdx.x] = v;
}

__global__ void k_relu(float* __restrict__ h) {
    int i = blockIdx.x * 256 + threadIdx.x;
    if (i < NNODES * D) h[i] = fmaxf(h[i], 0.f);
}

// ---------------------------------------------------------------------------
// Head: out[i, j] = hp[i, :] . Wp2[:, j] + bp2[j]   (Wp2 is [128 x 3])
// ---------------------------------------------------------------------------
__global__ void k_head(const float* __restrict__ hp, const float* __restrict__ Wp2,
                       const float* __restrict__ bp2, float* __restrict__ out)
{
    int idx = blockIdx.x * 256 + threadIdx.x;
    if (idx >= NNODES * DOUT) return;
    int node = idx / DOUT, j = idx - node * DOUT;
    const float* hr = hp + node * D;
    float s = bp2[j];
#pragma unroll 8
    for (int f = 0; f < D; ++f) s += hr[f] * Wp2[f * DOUT + j];
    out[idx] = s;
}

// ---------------------------------------------------------------------------
extern "C" void kernel_launch(void* const* d_in, const int* in_sizes, int n_in,
                              void* d_out, int out_size, void* d_ws, size_t ws_size,
                              hipStream_t stream) {
    const float* x    = (const float*)d_in[0];
    const int*   ei   = (const int*)  d_in[1];
    const float* W1   = (const float*)d_in[2];
    const float* b1   = (const float*)d_in[3];
    const float* W2   = (const float*)d_in[4];
    const float* b2   = (const float*)d_in[5];
    const float* W3   = (const float*)d_in[6];
    const float* b3   = (const float*)d_in[7];
    const float* g1   = (const float*)d_in[8];
    const float* be1  = (const float*)d_in[9];
    const float* g2   = (const float*)d_in[10];
    const float* be2  = (const float*)d_in[11];
    const float* Wp1  = (const float*)d_in[12];
    const float* bp1  = (const float*)d_in[13];
    const float* Wp2  = (const float*)d_in[14];
    const float* bp2  = (const float*)d_in[15];
    float* out = (float*)d_out;

    const int* row = ei;            // edge_index[0]
    const int* col = ei + NEDGES;   // edge_index[1]

    // workspace: dinv | 4 packed bf16 weights | two 50000x128 f32 buffers
    char* ws = (char*)d_ws;
    float* dinv = (float*)ws;
    size_t off = ((size_t)NNODES * sizeof(float) + 255) & ~(size_t)255;
    __bf16* pW1  = (__bf16*)(ws + off);
    __bf16* pW2  = pW1 + WPACK;
    __bf16* pW3  = pW2 + WPACK;
    __bf16* pWp1 = pW3 + WPACK;
    float* bufA = (float*)(ws + off + 4 * WPACK * sizeof(__bf16));
    float* bufB = bufA + (size_t)NNODES * D;

    const dim3 blk1(256);
    const dim3 blkW(32, 8);
    const int gN   = (NNODES + 255) / 256;          // node grid
    const int gE   = (NEDGES + 255) / 256;          // edge grid
    const int gF   = (NNODES * D) / 256;            // feature-elem grid (25000)
    const int gT   = NNODES / 16;                   // 3125 WMMA tiles
    const int gEw  = (NEDGES + 7) / 8;              // 1 wave per edge (200000)
    const int gNw  = (NNODES + 7) / 8;              // 1 wave per node
    const int gO   = (NNODES * DOUT + 255) / 256;
    const int gP   = WPACK / 256;                   // 64 blocks per weight pack

    // one-time bf16 weight packing (fragment-ordered)
    k_pack_w<<<gP, blk1, 0, stream>>>(W1,  pW1);
    k_pack_w<<<gP, blk1, 0, stream>>>(W2,  pW2);
    k_pack_w<<<gP, blk1, 0, stream>>>(W3,  pW3);
    k_pack_w<<<gP, blk1, 0, stream>>>(Wp1, pWp1);

    // symmetric GCN normalization
    k_deg_init <<<gN, blk1, 0, stream>>>(dinv);
    k_deg_accum<<<gE, blk1, 0, stream>>>(col, dinv);
    k_deg_rsqrt<<<gN, blk1, 0, stream>>>(dinv);

    // layer 0
    k_gemm_bf16<<<gT,  blkW, 0, stream>>>(x, pW1, nullptr, bufA);
    k_agg_init <<<gF,  blk1, 0, stream>>>(bufA, b1, dinv, bufB);
    k_scatter  <<<gEw, blkW, 0, stream>>>(bufA, row, col, dinv, bufB);
    k_relu_ln  <<<gNw, blkW, 0, stream>>>(bufB, g1, be1);

    // layer 1
    k_gemm_bf16<<<gT,  blkW, 0, stream>>>(bufB, pW2, nullptr, bufA);
    k_agg_init <<<gF,  blk1, 0, stream>>>(bufA, b2, dinv, bufB);
    k_scatter  <<<gEw, blkW, 0, stream>>>(bufA, row, col, dinv, bufB);
    k_relu_ln  <<<gNw, blkW, 0, stream>>>(bufB, g2, be2);

    // layer 2 (relu only)
    k_gemm_bf16<<<gT,  blkW, 0, stream>>>(bufB, pW3, nullptr, bufA);
    k_agg_init <<<gF,  blk1, 0, stream>>>(bufA, b3, dinv, bufB);
    k_scatter  <<<gEw, blkW, 0, stream>>>(bufA, row, col, dinv, bufB);
    k_relu     <<<gF,  blk1, 0, stream>>>(bufB);

    // MLP head
    k_gemm_bf16<<<gT,  blkW, 0, stream>>>(bufB, pWp1, bp1, bufA);
    k_head     <<<gO,  blk1, 0, stream>>>(bufA, Wp2, bp2, out);
}
